// VectorQuantizer_35399120454175
// MI455X (gfx1250) — matrix operand order repaired and checked by reading, not compiled, not run on hardware
//
#include <hip/hip_runtime.h>
#include <hip/hip_bf16.h>

typedef __attribute__((ext_vector_type(16))) __bf16 v16bf;
typedef __attribute__((ext_vector_type(8)))  __bf16 v8bf;
typedef __attribute__((ext_vector_type(8)))  float  v8f;

#define CDIM 256
#define BATCH 32
#define HDIM 64
#define WDIM 64
#define KCODES 1024
#define NPIX (BATCH*HDIM*WDIM)                       // 131072
#define ZQ_ELEMS ((size_t)BATCH*CDIM*HDIM*WDIM)      // 33554432
#define NBLK (BATCH*HDIM)                            // 2048

static constexpr size_t VQ_OFF   = ZQ_ELEMS;
static constexpr size_t IDX_OFF  = VQ_OFF + 1;
static constexpr size_t TOPO_OFF = IDX_OFF + (size_t)NPIX;

// ---------------- Kernel 1: normalize codebook + pack B fragments ----------------
// norm_w (f32, [1024][256]) and bpack (bf16) in WMMA-B fragment order:
// fragment for (code-tile ct, k-step ks): 32 lanes x 16 halves contiguous.
// lane slot L: code row = ct*16 + (L%16); channels = ks*32 + (L<16?0:16) + j, j=0..15.
__global__ void vq_norm_pack_kernel(const float* __restrict__ w,
                                    float* __restrict__ nw,
                                    __bf16* __restrict__ bpack) {
    __shared__ float row[8][CDIM];
    int wv = threadIdx.x >> 5, lane = threadIdx.x & 31;
    int code = blockIdx.x * 8 + wv;
    const float* src = w + (size_t)code * CDIM;
    float vals[8]; float s = 0.f;
    #pragma unroll
    for (int i = 0; i < 8; ++i) { float v = src[lane + 32*i]; vals[i] = v; s += v*v; }
    #pragma unroll
    for (int m = 16; m >= 1; m >>= 1) s += __shfl_xor(s, m, 32);
    float scale = 1.0f / fmaxf(sqrtf(s), 1e-12f);
    #pragma unroll
    for (int i = 0; i < 8; ++i) {
        float v = vals[i] * scale;
        nw[(size_t)code*CDIM + lane + 32*i] = v;
        row[wv][lane + 32*i] = v;
    }
    __syncthreads();
    int ct = code >> 4, cl = code & 15;
    int half = (lane >> 4) & 1, j = lane & 15;
    #pragma unroll
    for (int ks = 0; ks < 8; ++ks) {
        float v = row[wv][ks*32 + half*16 + j];
        bpack[(((size_t)(ct*8 + ks)*32) + cl + half*16)*16 + j] = (__bf16)v;
    }
}

// ---------------- Kernel 2: topology grid smoothness loss ----------------
__global__ void vq_topo_kernel(const float* __restrict__ nw, float* __restrict__ out_topo) {
    int tid = threadIdx.x;
    float a1 = 0.f, a2 = 0.f, a3 = 0.f, a4 = 0.f;
    for (int p = tid; p < 32*31*256; p += 256) {
        int c = p & 255; int rest = p >> 8; int j = rest % 31 + 1; int i = rest / 31;
        float d = nw[(i*32 + j)*256 + c] - nw[(i*32 + j - 1)*256 + c]; a1 += d*d;
    }
    for (int p = tid; p < 31*32*256; p += 256) {
        int c = p & 255; int rest = p >> 8; int j = rest & 31; int i = (rest >> 5) + 1;
        float d = nw[(i*32 + j)*256 + c] - nw[((i-1)*32 + j)*256 + c]; a2 += d*d;
    }
    for (int p = tid; p < 32*256; p += 256) {
        int c = p & 255; int i = p >> 8;
        float d = nw[(i*32 + 0)*256 + c] - nw[(i*32 + 31)*256 + c]; a3 += d*d;
    }
    for (int p = tid; p < 32*256; p += 256) {
        int c = p & 255; int j = p >> 8;
        float d = nw[j*256 + c] - nw[(31*32 + j)*256 + c]; a4 += d*d;
    }
    const float inv1 = 1.0f / (32.0f*31.0f*256.0f);
    const float inv3 = 1.0f / (32.0f*256.0f);
    float acc = (a1 + a2)*inv1 + (a3 + a4)*inv3;
    __shared__ float red[256];
    red[tid] = acc; __syncthreads();
    for (int k = 128; k >= 1; k >>= 1) { if (tid < k) red[tid] += red[tid + k]; __syncthreads(); }
    if (tid == 0) *out_topo = red[0] * 0.35f;
}

// ---------------- Kernel 3: main VQ (normalize + WMMA GEMM argmax + outputs) ----------------
__global__ __launch_bounds__(128) void vq_main_kernel(const float* __restrict__ ze,
                                                      const float* __restrict__ nw,
                                                      const __bf16* __restrict__ bpack,
                                                      float* __restrict__ dout,
                                                      float* __restrict__ partials) {
    __shared__ __align__(16) __bf16 lds_a[64*264];    // [pixel][channel], padded pitch 264
    __shared__ __align__(16) __bf16 lds_bb[2][4096];  // double-buffered B tile (2 x 8KB)
    __shared__ float lds_scale[64];
    __shared__ float lds_s[2][64];
    __shared__ int   lds_idx[64];
    __shared__ float lds_red[128];

    int tid = threadIdx.x;
    int bh = blockIdx.x;                 // b*64 + h
    int b = bh >> 6, h = bh & 63;
    size_t inbase = (size_t)b*(CDIM*HDIM*WDIM) + (size_t)h*WDIM;

    // ---- pass 1: per-pixel sum of squares ----
    {
        int wpx = tid & 63, half = tid >> 6;
        float s = 0.f;
        for (int c = half*128; c < half*128 + 128; ++c) {
            float v = ze[inbase + (size_t)c*4096 + wpx];
            s += v*v;
        }
        lds_s[half][wpx] = s;
    }
    __syncthreads();
    if (tid < 64) {
        float s = lds_s[0][tid] + lds_s[1][tid];
        lds_scale[tid] = 1.0f / fmaxf(sqrtf(s), 1e-12f);
    }
    __syncthreads();
    // ---- pass 2: normalized bf16 A-tile into LDS (re-read hits cache) ----
    {
        int wpx = tid & 63, half = tid >> 6;
        float sc = lds_scale[wpx];
        for (int c = half*128; c < half*128 + 128; ++c) {
            float v = ze[inbase + (size_t)c*4096 + wpx] * sc;
            lds_a[wpx*264 + c] = (__bf16)v;
        }
    }
    __syncthreads();

    // ---- WMMA GEMM + running argmax ----
    int wv = tid >> 5, lane = tid & 31;
    int m = lane & 15;
    int kbase = (lane < 16) ? 0 : 16;    // A-matrix 16x32 bf16, contiguous k = 16h + s (matches bpack)
    int mglob = wv*16 + m;
    const __bf16* arow = lds_a + mglob*264;

    v16bf afrag[8];
    #pragma unroll
    for (int ks = 0; ks < 8; ++ks) {
        v8bf lo = *(const v8bf*)(arow + ks*32 + kbase);
        v8bf hi = *(const v8bf*)(arow + ks*32 + kbase + 8);
        v16bf a;
        #pragma unroll
        for (int j = 0; j < 8; ++j) { a[j] = lo[j]; a[j+8] = hi[j]; }
        afrag[ks] = a;
    }

    float bestVal[8]; int bestIdx[8];
    #pragma unroll
    for (int r = 0; r < 8; ++r) { bestVal[r] = -1e30f; bestIdx[r] = 0; }

    // LDS raw offsets for async-copy destinations (shared aperture keeps the
    // LDS byte offset in the low 32 bits of the flat address on gfx1250)
    unsigned lbb0 = (unsigned)(uintptr_t)(void*)&lds_bb[0][0] + (unsigned)tid*16u;
    unsigned lbb1 = (unsigned)(uintptr_t)(void*)&lds_bb[1][0] + (unsigned)tid*16u;

    // prologue: async-stage code tile 0 into buffer 0 (4 x b128 per thread = 8KB/WG)
    {
        const char* gsrc = (const char*)bpack + (size_t)tid*16;
        #pragma unroll
        for (int j = 0; j < 4; ++j) {
            asm volatile("global_load_async_to_lds_b128 %0, %1, off"
                         :: "v"(lbb0 + (unsigned)j*2048u), "v"(gsrc + (size_t)j*2048)
                         : "memory");
        }
        asm volatile("s_wait_asynccnt 0x0" ::: "memory");
    }
    __syncthreads();

    for (int ct = 0; ct < 64; ++ct) {
        int cur = ct & 1;
        // prefetch next tile into the other buffer (free: barrier at loop tail)
        if (ct < 63) {
            const char* gsrc = (const char*)bpack + (size_t)(ct + 1)*8192 + (size_t)tid*16;
            unsigned ldst = cur ? lbb0 : lbb1;
            #pragma unroll
            for (int j = 0; j < 4; ++j) {
                asm volatile("global_load_async_to_lds_b128 %0, %1, off"
                             :: "v"(ldst + (unsigned)j*2048u), "v"(gsrc + (size_t)j*2048)
                             : "memory");
            }
        }
        // compute: two accumulator chains to break WMMA C-operand serialization
        const __bf16* bbase = &lds_bb[cur][0] + lane*16;
        v8f acc0 = {}, acc1 = {};
        #pragma unroll
        for (int ks = 0; ks < 8; ks += 2) {
            v16bf b0 = *(const v16bf*)(bbase + (ks + 0)*512);
            v16bf b1 = *(const v16bf*)(bbase + (ks + 1)*512);
            acc0 = __builtin_amdgcn_wmma_f32_16x16x32_bf16(
                       false, afrag[ks + 0], false, b0, (short)0, acc0, false, false);
            acc1 = __builtin_amdgcn_wmma_f32_16x16x32_bf16(
                       false, afrag[ks + 1], false, b1, (short)0, acc1, false, false);
        }
        int code = ct*16 + m;
        #pragma unroll
        for (int r = 0; r < 8; ++r) {
            float v = acc0[r] + acc1[r];
            if (v > bestVal[r]) { bestVal[r] = v; bestIdx[r] = code; }
        }
        if (ct < 63) asm volatile("s_wait_asynccnt 0x0" ::: "memory");
        __syncthreads();
    }

    // 16-lane butterfly reduction (lanes 0-15 = pixel row r, lanes 16-31 = row r+8)
    #pragma unroll
    for (int r = 0; r < 8; ++r) {
        float v = bestVal[r]; int ix = bestIdx[r];
        #pragma unroll
        for (int msk = 8; msk >= 1; msk >>= 1) {
            float ov = __shfl_xor(v, msk, 32);
            int   oi = __shfl_xor(ix, msk, 32);
            if (ov > v || (ov == v && oi < ix)) { v = ov; ix = oi; }
        }
        bestVal[r] = v; bestIdx[r] = ix;
    }
    if (lane == 0) {
        #pragma unroll
        for (int r = 0; r < 8; ++r) lds_idx[wv*16 + r] = bestIdx[r];
    }
    if (lane == 16) {
        #pragma unroll
        for (int r = 0; r < 8; ++r) lds_idx[wv*16 + 8 + r] = bestIdx[r];
    }
    __syncthreads();

    // ---- outputs: idx, z_q (BCHW), loss partial ----
    if (tid < 64) {
        dout[IDX_OFF + (size_t)bh*64 + tid] = (float)lds_idx[tid];
    }
    float local = 0.f;
    for (int j = tid; j < CDIM*WDIM; j += 128) {
        int c = j >> 6, wpx = j & 63;
        int code = lds_idx[wpx];
        float q = nw[(size_t)code*CDIM + c];
        size_t off = inbase + (size_t)c*4096 + wpx;
        float xen = ze[off] * lds_scale[wpx];
        float d = q - xen;
        local += d*d;
        dout[off] = q;                    // BCHW z_q index == z_e index
    }
    lds_red[tid] = local;
    __syncthreads();
    for (int s = 64; s >= 1; s >>= 1) { if (tid < s) lds_red[tid] += lds_red[tid + s]; __syncthreads(); }
    if (tid == 0) partials[bh] = lds_red[0];
}

// ---------------- Kernel 4: deterministic loss finalize ----------------
__global__ void vq_finalize_kernel(const float* __restrict__ partials, float* __restrict__ dout) {
    __shared__ float red[256];
    int tid = threadIdx.x;
    float s = 0.f;
    for (int i = tid; i < NBLK; i += 256) s += partials[i];
    red[tid] = s; __syncthreads();
    for (int k = 128; k >= 1; k >>= 1) { if (tid < k) red[tid] += red[tid + k]; __syncthreads(); }
    if (tid == 0) dout[VQ_OFF] = 1.25f * red[0] / (float)ZQ_ELEMS;
}

extern "C" void kernel_launch(void* const* d_in, const int* in_sizes, int n_in,
                              void* d_out, int out_size, void* d_ws, size_t ws_size,
                              hipStream_t stream) {
    const float* ze = (const float*)d_in[0];    // (32,256,64,64) f32
    const float* w  = (const float*)d_in[1];    // (1024,256) f32
    float* out = (float*)d_out;

    // workspace layout
    float*  nw       = (float*)d_ws;                                     // 1 MB
    __bf16* bpack    = (__bf16*)((char*)d_ws + (size_t)1048576);         // 512 KB
    float*  partials = (float*)((char*)d_ws + (size_t)1048576 + 524288); // 8 KB

    vq_norm_pack_kernel<<<128, 256, 0, stream>>>(w, nw, bpack);
    vq_topo_kernel<<<1, 256, 0, stream>>>(nw, out + TOPO_OFF);
    vq_main_kernel<<<NBLK, 128, 0, stream>>>(ze, nw, bpack, out, partials);
    vq_finalize_kernel<<<1, 256, 0, stream>>>(partials, out);
}
